// ParallelGroupedMLP_40553081209075
// MI455X (gfx1250) — matrix operand order repaired and checked by reading, not compiled, not run on hardware
//
#include <hip/hip_runtime.h>
#include <hip/hip_bf16.h>
#include <stdint.h>

typedef _Float16 v16h __attribute__((ext_vector_type(16)));
typedef _Float16 v8h  __attribute__((ext_vector_type(8)));
typedef _Float16 v4h  __attribute__((ext_vector_type(4)));
typedef float    v8f  __attribute__((ext_vector_type(8)));

#define NE 8
#define HD 1024
#define FF 4096
#define TT 32768
#define GG (TT / NE)     // 4096 tokens per expert

#define BM 128           // block tile M
#define BN 256           // block tile N
#define BK 32            // K step (one WMMA K)
#define LSTR 40          // padded LDS row stride in halves (80B rows: 16B-aligned, conflict-free)

// A fragment (16x32 f16, MxK): lane l -> row m0 + (l&15); K halves [kk0,kk0+8) and [kk0+16,kk0+24)
__device__ __forceinline__ v16h load_a_frag(const _Float16* s, int lane, int m0) {
    const int row = m0 + (lane & 15);
    const int kk0 = (lane < 16) ? 0 : 8;
    const _Float16* p = s + row * LSTR + kk0;
    v8h lo = *(const v8h*)(p);
    v8h hi = *(const v8h*)(p + 16);
    return __builtin_shufflevector(lo, hi, 0,1,2,3,4,5,6,7,8,9,10,11,12,13,14,15);
}

// B fragment (32x16 f16, KxN): lane l -> col n0 + (l&15); K halves [k0,k0+8) and [k0+16,k0+24)
__device__ __forceinline__ v16h load_b_frag(const _Float16* s, int lane, int n0) {
    const int col = n0 + (lane & 15);
    const int k0  = (lane < 16) ? 0 : 8;
    const _Float16* p = s + col * LSTR + k0;
    v8h lo = *(const v8h*)(p);
    v8h hi = *(const v8h*)(p + 16);
    return __builtin_shufflevector(lo, hi, 0,1,2,3,4,5,6,7,8,9,10,11,12,13,14,15);
}

// Per-lane async global->LDS copy of 16 bytes (ASYNCcnt-tracked, no VGPR staging).
__device__ __forceinline__ void async_cp16(uint32_t lds_off, const _Float16* gptr) {
    asm volatile("global_load_async_to_lds_b128 %0, %1, off"
                 :: "v"(lds_off), "v"(gptr) : "memory");
}

__device__ __forceinline__ void wait_async0() {
    asm volatile("s_wait_asynccnt 0" ::: "memory");
}

// ---------------- unified f16 NT GEMM: C = [relu](A @ B^T) per expert group ----------------
// A: [T, KDIM] f16 row-major. B: [E*NPE, KDIM] f16 row-major (rows = output cols, per expert).
// C: [T, ldc] (OUT_T). Block: 128x256, 8 waves, 64x64 per wave, double-buffered async LDS.
template <int KDIM, int NPE, bool RELU, typename OUT_T>
__global__ __launch_bounds__(256) void gemm_nt_f16(
    const _Float16* __restrict__ A, const _Float16* __restrict__ B,
    OUT_T* __restrict__ C, const int ldc)
{
    __shared__ _Float16 sA[2][BM * LSTR];
    __shared__ _Float16 sB[2][BN * LSTR];

    const int tid  = threadIdx.x;
    const int lane = tid & 31;
    const int wave = tid >> 5;
    const int wm = wave >> 2;              // 0..1 : 64-row slab
    const int wn = wave & 3;               // 0..3 : 64-col slab

    const int mBase = blockIdx.x * BM;
    const int nBase = blockIdx.y * BN;
    const int e = mBase / GG;              // expert id (BM divides GG)
    const int bRow0 = e * NPE + nBase;

    const uint32_t sA0 = (uint32_t)(uintptr_t)&sA[0][0];
    const uint32_t sA1 = (uint32_t)(uintptr_t)&sA[1][0];
    const uint32_t sB0 = (uint32_t)(uintptr_t)&sB[0][0];
    const uint32_t sB1 = (uint32_t)(uintptr_t)&sB[1][0];

    // Issue all async copies for one K-step into buffer `buf`.
    // A tile: 128 rows x 64B (512 x 16B chunks, 2 per wave). B tile: 256 rows (4 per wave).
    auto issue = [&](int buf, int kb) {
        const uint32_t aBase = buf ? sA1 : sA0;
        const uint32_t bBase = buf ? sB1 : sB0;
#pragma unroll
        for (int i = 0; i < 2; ++i) {
            const int c = wave * 64 + i * 32 + lane;
            const int r = c >> 2, kc = c & 3;
            async_cp16(aBase + (uint32_t)(r * (LSTR * 2) + kc * 16),
                       A + (mBase + r) * KDIM + kb + kc * 8);
        }
#pragma unroll
        for (int i = 0; i < 4; ++i) {
            const int c = wave * 128 + i * 32 + lane;
            const int r = c >> 2, kc = c & 3;
            async_cp16(bBase + (uint32_t)(r * (LSTR * 2) + kc * 16),
                       B + (bRow0 + r) * KDIM + kb + kc * 8);
        }
    };

    v8f acc[4][4];
#pragma unroll
    for (int mi = 0; mi < 4; ++mi)
#pragma unroll
        for (int ni = 0; ni < 4; ++ni)
#pragma unroll
            for (int r = 0; r < 8; ++r) acc[mi][ni][r] = 0.0f;

    issue(0, 0);
    wait_async0();
    __syncthreads();

    for (int kb = 0; kb < KDIM; kb += BK) {
        const int buf = (kb >> 5) & 1;
        if (kb + BK < KDIM) issue(buf ^ 1, kb + BK);   // prefetch next tile into other buffer

        const _Float16* a_s = sA[buf];
        const _Float16* b_s = sB[buf];
        v16h a[4], b[4];
#pragma unroll
        for (int mi = 0; mi < 4; ++mi) a[mi] = load_a_frag(a_s, lane, wm * 64 + mi * 16);
#pragma unroll
        for (int ni = 0; ni < 4; ++ni) b[ni] = load_b_frag(b_s, lane, wn * 64 + ni * 16);
#pragma unroll
        for (int mi = 0; mi < 4; ++mi)
#pragma unroll
            for (int ni = 0; ni < 4; ++ni)
                acc[mi][ni] = __builtin_amdgcn_wmma_f32_16x16x32_f16(
                    false, a[mi], false, b[ni], (short)0, acc[mi][ni], false, false);

        wait_async0();        // next buffer's data (our portion) has landed
        __syncthreads();      // everyone's portion landed + everyone done reading cur buffer
    }

    // C/D layout: VGPR r, lane l -> M = r + (l/16)*8, N = l%16
#pragma unroll
    for (int mi = 0; mi < 4; ++mi)
#pragma unroll
        for (int ni = 0; ni < 4; ++ni) {
            const int row0 = mBase + wm * 64 + mi * 16 + (lane >> 4) * 8;
            const int col  = nBase + wn * 64 + ni * 16 + (lane & 15);
#pragma unroll
            for (int r = 0; r < 8; ++r) {
                float v = acc[mi][ni][r];
                if (RELU) v = v > 0.0f ? v : 0.0f;
                C[(row0 + r) * ldc + col] = (OUT_T)v;
            }
        }
}

// ---------------- prep kernels (memory-bound) ----------------
__global__ __launch_bounds__(256) void cvt_f16_kernel(const float4* __restrict__ in,
                                                      v4h* __restrict__ out) {
    const int i = blockIdx.x * 256 + threadIdx.x;
    float4 f = in[i];
    v4h h = { (_Float16)f.x, (_Float16)f.y, (_Float16)f.z, (_Float16)f.w };
    out[i] = h;
}

// w2t[e][n][f] = (f16) w2[e][f][n]  -> GEMM2 becomes NT with K=FF contiguous
__global__ __launch_bounds__(256) void transpose_cvt_w2(const float* __restrict__ w2,
                                                        _Float16* __restrict__ w2t) {
    __shared__ _Float16 t[32][34];
    const int f0 = blockIdx.x * 32;
    const int n0 = blockIdx.y * 32;
    const int e  = blockIdx.z;
    const float* src = w2 + (size_t)e * FF * HD;
    _Float16* dst    = w2t + (size_t)e * HD * FF;

    const int r  = threadIdx.x >> 3;   // 0..31
    const int c4 = threadIdx.x & 7;    // 0..7
    float4 v = *(const float4*)(src + (f0 + r) * HD + n0 + c4 * 4);
    t[c4 * 4 + 0][r] = (_Float16)v.x;
    t[c4 * 4 + 1][r] = (_Float16)v.y;
    t[c4 * 4 + 2][r] = (_Float16)v.z;
    t[c4 * 4 + 3][r] = (_Float16)v.w;
    __syncthreads();
    _Float16* d = dst + (n0 + r) * FF + f0 + c4 * 4;
    d[0] = t[r][c4 * 4 + 0];
    d[1] = t[r][c4 * 4 + 1];
    d[2] = t[r][c4 * 4 + 2];
    d[3] = t[r][c4 * 4 + 3];
}

extern "C" void kernel_launch(void* const* d_in, const int* in_sizes, int n_in,
                              void* d_out, int out_size, void* d_ws, size_t ws_size,
                              hipStream_t stream) {
    const float* x  = (const float*)d_in[0];   // [T, H]
    const float* w1 = (const float*)d_in[1];   // [E*FF, H]
    const float* w2 = (const float*)d_in[2];   // [E*FF, H]
    float* out = (float*)d_out;                // [T, H] f32

    char* ws = (char*)d_ws;
    _Float16* hid = (_Float16*)ws;                                  // [T, FF]   256 MB
    _Float16* xh  = (_Float16*)(ws + (size_t)TT * FF * 2);          // [T, H]     64 MB
    _Float16* w1h = xh + (size_t)TT * HD;                           // [E*FF, H]  64 MB
    _Float16* w2t = w1h + (size_t)NE * FF * HD;                     // [E*H, FF]  64 MB

    dim3 blk(256);
    cvt_f16_kernel<<<dim3((TT * HD / 4) / 256), blk, 0, stream>>>((const float4*)x, (v4h*)xh);
    cvt_f16_kernel<<<dim3((NE * FF * HD / 4) / 256), blk, 0, stream>>>((const float4*)w1, (v4h*)w1h);
    transpose_cvt_w2<<<dim3(FF / 32, HD / 32, NE), blk, 0, stream>>>(w2, w2t);

    // GEMM1: hid = relu(xh @ w1h^T)  [K=1024, N per expert = FF]
    gemm_nt_f16<HD, FF, true, _Float16>
        <<<dim3(TT / BM, FF / BN), blk, 0, stream>>>(xh, w1h, hid, FF);
    // GEMM2: out = hid @ w2t^T       [K=4096, N per expert = H]
    gemm_nt_f16<FF, HD, false, float>
        <<<dim3(TT / BM, HD / BN), blk, 0, stream>>>(hid, w2t, out, HD);
}
